// WMGPT_3315714753062
// MI455X (gfx1250) — matrix operand order repaired and checked by reading, not compiled, not run on hardware
//
#include <hip/hip_runtime.h>
#include <hip/hip_bf16.h>
#include <cstdint>

// ---------------------------------------------------------------------------
// WMGPT forward for MI455X (gfx1250, wave32, WMMA).
// ~413 GFLOP/forward; bf16 WMMA (16x16x32, f32 accum) for all GEMMs,
// flash-attention streaming (no 256MB score matrix), f32 LayerNorm/softmax.
// GEMM: double-buffered LDS, A tiles via GLOBAL_LOAD_ASYNC_TO_LDS_B128
// (ASYNCcnt), B tiles register-pipelined with swizzled ds stores.
// All register-array loops force-unrolled so nothing is demoted to scratch.
// ---------------------------------------------------------------------------

typedef __attribute__((ext_vector_type(16))) __bf16 v16bf;
typedef __attribute__((ext_vector_type(8)))  float  v8f;

#define BB   8
#define TT   1024
#define CC   512
#define HH   8
#define KDIM 64
#define FFD  2048
#define LL   6
#define NTOK (BB * TT)   // 8192

__device__ __forceinline__ unsigned short f2bfu(float f) {
  unsigned u = __float_as_uint(f);
  u += 0x7FFFu + ((u >> 16) & 1u);          // round-to-nearest-even
  return (unsigned short)(u >> 16);
}
__device__ __forceinline__ __bf16 f2bf(float f) {
  unsigned short s = f2bfu(f);
  return __builtin_bit_cast(__bf16, s);
}

union Frag { v16bf v; uint4 q[2]; };

// Async copy of 16 bytes per lane: global -> LDS (tracked by ASYNCcnt).
__device__ __forceinline__ void async_g2l_b128(void* lds, const void* gptr) {
  unsigned loff = (unsigned)(uintptr_t)lds;                 // low 32b = LDS offset
  unsigned long long ga = (unsigned long long)(uintptr_t)gptr;
  asm volatile("global_load_async_to_lds_b128 %0, %1, off"
               :: "v"(loff), "v"(ga) : "memory");
}
__device__ __forceinline__ void wait_asynccnt0() {
  asm volatile("s_wait_asynccnt 0x0" ::: "memory");
}

// ---------------------------------------------------------------------------
// Generic GEMM: C[M,N] = A[M,K](bf16) * B[K,N](bf16) (+bias)(+relu)
// Macro tile 128x128, 256 threads = 8 waves (4 in M x 2 in N),
// each wave -> 32x64 = 2x4 WMMA accumulators. K step 32. Double buffered.
// ---------------------------------------------------------------------------
__global__ __launch_bounds__(256, 1) void k_gemm(
    const __bf16* __restrict__ A, int lda,
    const __bf16* __restrict__ B, int ldb,
    float* __restrict__ outF, int ldf,
    __bf16* __restrict__ outB, int ldob,
    const float* __restrict__ bias, int relu, int Kdim)
{
  __shared__ __align__(16) unsigned short shA[2][128 * 40];  // row stride 40 (pad)
  __shared__ __align__(16) unsigned short shB[2][4096];      // fragment-swizzled

  const int tid  = threadIdx.x;
  const int lane = tid & 31;
  const int w    = tid >> 5;
  const int wm   = w >> 1;              // 0..3
  const int wn   = w & 1;               // 0..1
  const int tm0  = blockIdx.y * 128;
  const int tn0  = blockIdx.x * 128;
  const int lg   = lane >> 4;           // lane half
  const int ll   = lane & 15;

  // fixed per-thread staging coordinates
  const int lin0 = tid * 2, lin1 = tid * 2 + 1;
  const int am0 = lin0 >> 2, akq0 = (lin0 & 3) * 8;   // A: [m 0..127][kq 0/8/16/24]
  const int am1 = lin1 >> 2, akq1 = (lin1 & 3) * 8;
  const int bk0 = lin0 >> 4, bnq0 = (lin0 & 15) * 8;  // B: [k 0..31][nq 0..120]
  const int bk1 = lin1 >> 4, bnq1 = (lin1 & 15) * 8;

  v8f acc[2][4];
#pragma unroll
  for (int i = 0; i < 2; i++)
#pragma unroll
    for (int j = 0; j < 4; j++)
#pragma unroll
      for (int e = 0; e < 8; e++) acc[i][j][e] = 0.0f;

  auto issueA = [&](int buf, int kb) {
    async_g2l_b128(&shA[buf][am0 * 40 + akq0],
                   A + (size_t)(tm0 + am0) * lda + kb + akq0);
    async_g2l_b128(&shA[buf][am1 * 40 + akq1],
                   A + (size_t)(tm0 + am1) * lda + kb + akq1);
  };
  auto storeB = [&](int buf, uint4 v, int k, int nq) {
    const unsigned short* e = (const unsigned short*)&v;
    int khi = k >> 4, kidx = k & 15;
#pragma unroll
    for (int j = 0; j < 8; j++) {
      int n = nq + j;
      shB[buf][(n >> 4) * 512 + ((n & 15) + 16 * khi) * 16 + kidx] = e[j];
    }
  };

  // ---- prologue: stage tile kb=0 into buffer 0 ----
  {
    uint4 b0 = *(const uint4*)(B + (size_t)bk0 * ldb + tn0 + bnq0);
    uint4 b1 = *(const uint4*)(B + (size_t)bk1 * ldb + tn0 + bnq1);
    issueA(0, 0);
    storeB(0, b0, bk0, bnq0);
    storeB(0, b1, bk1, bnq1);
    wait_asynccnt0();
    __syncthreads();
  }

  int cur = 0;
  for (int kb = 0; kb < Kdim; kb += 32) {
    const bool hasNext = (kb + 32) < Kdim;
    uint4 nb0, nb1;
    if (hasNext) {   // issue next-tile global traffic before the math
      nb0 = *(const uint4*)(B + (size_t)(kb + 32 + bk0) * ldb + tn0 + bnq0);
      nb1 = *(const uint4*)(B + (size_t)(kb + 32 + bk1) * ldb + tn0 + bnq1);
      issueA(cur ^ 1, kb + 32);
    }
    // ---- fragments + WMMA from current buffer ----
    Frag a[2], b[4];
#pragma unroll
    for (int si = 0; si < 2; si++) {
      int m = wm * 32 + si * 16 + ll;
      const unsigned short* p = &shA[cur][m * 40 + 16 * lg];
      a[si].q[0] = *(const uint4*)p;
      a[si].q[1] = *(const uint4*)(p + 8);
    }
#pragma unroll
    for (int ni = 0; ni < 4; ni++) {
      int nb = wn * 4 + ni;
      const unsigned short* p = &shB[cur][nb * 512 + lane * 16];
      b[ni].q[0] = *(const uint4*)p;
      b[ni].q[1] = *(const uint4*)(p + 8);
    }
#pragma unroll
    for (int si = 0; si < 2; si++)
#pragma unroll
      for (int ni = 0; ni < 4; ni++)
        acc[si][ni] = __builtin_amdgcn_wmma_f32_16x16x32_bf16(
            false, a[si].v, false, b[ni].v, (short)0, acc[si][ni], false, false);
    if (hasNext) {
      storeB(cur ^ 1, nb0, bk0, bnq0);
      storeB(cur ^ 1, nb1, bk1, bnq1);
      wait_asynccnt0();
    }
    __syncthreads();
    cur ^= 1;
  }

  // ---- epilogue: bias / relu / dual-precision store (fully unrolled so the
  //      accumulator array is never dynamically indexed -> stays in VGPRs) ----
#pragma unroll
  for (int si = 0; si < 2; si++) {
    int mrow0 = tm0 + wm * 32 + si * 16 + 8 * lg;
#pragma unroll
    for (int ni = 0; ni < 4; ni++) {
      int col = tn0 + wn * 64 + ni * 16 + ll;
      float bv = bias ? bias[col] : 0.0f;
#pragma unroll
      for (int r = 0; r < 8; r++) {
        float v = acc[si][ni][r] + bv;
        if (relu) v = fmaxf(v, 0.0f);
        int row = mrow0 + r;
        if (outF) outF[(size_t)row * ldf + col] = v;
        if (outB) outB[(size_t)row * ldob + col] = f2bf(v);
      }
    }
  }
}

// ---------------------------------------------------------------------------
// Flash attention. Grid (T/64, H, B), 128 threads = 4 waves, 16 queries/wave.
// S = QK^T via WMMA (K from global, per-lane contiguous), online softmax in
// registers, P->LDS->A-fragment, O += P*V with V staged in LDS.
// ---------------------------------------------------------------------------
__global__ __launch_bounds__(128, 1) void k_attn(
    const __bf16* __restrict__ Qm, const __bf16* __restrict__ Km,
    const __bf16* __restrict__ Vm, __bf16* __restrict__ Om)
{
  __shared__ __align__(16) unsigned short shV[2048];  // 32x64 bf16, frag order
  __shared__ __align__(16) unsigned short shP[2048];  // per-wave 16x32 bf16

  const int b    = blockIdx.z;
  const int h    = blockIdx.y;
  const int qt0  = blockIdx.x * 64;
  const int tid  = threadIdx.x;
  const int lane = tid & 31;
  const int w    = tid >> 5;
  const int lg   = lane >> 4;
  const int ll   = lane & 15;
  const int q0   = qt0 + w * 16;

  // Q fragments (16x64 -> 2 A fragments), direct from global
  Frag aq[2];
  {
    size_t rowbase = ((size_t)(b * TT + q0 + ll)) * CC + h * KDIM;
#pragma unroll
    for (int ks = 0; ks < 2; ks++) {
      const __bf16* p = Qm + rowbase + ks * 32 + 16 * lg;
      aq[ks].q[0] = *(const uint4*)p;
      aq[ks].q[1] = *(const uint4*)(p + 8);
    }
  }

  v8f o[4];
#pragma unroll
  for (int ni = 0; ni < 4; ni++)
#pragma unroll
    for (int e = 0; e < 8; e++) o[ni][e] = 0.0f;
  float mrow[8], lrow[8];
#pragma unroll
  for (int r = 0; r < 8; r++) { mrow[r] = -1e30f; lrow[r] = 0.0f; }

  for (int s0 = 0; s0 < qt0 + 64; s0 += 32) {
    // ---- cooperatively stage V[32 x 64] into LDS (fragment order) ----
    {
      int k  = tid >> 2;                // 0..31
      int ni = tid & 3;                 // 0..3
      const __bf16* src = Vm + ((size_t)(b * TT + s0 + k)) * CC + h * KDIM + ni * 16;
      uint4 v0 = *(const uint4*)src;
      uint4 v1 = *(const uint4*)(src + 8);
      const unsigned short* e0 = (const unsigned short*)&v0;
      const unsigned short* e1 = (const unsigned short*)&v1;
      int khi = k >> 4, kidx = k & 15;
#pragma unroll
      for (int j = 0; j < 8; j++)
        shV[ni * 512 + (j + 16 * khi) * 16 + kidx] = e0[j];
#pragma unroll
      for (int j = 0; j < 8; j++)
        shV[ni * 512 + ((8 + j) + 16 * khi) * 16 + kidx] = e1[j];
    }
    __syncthreads();

    if (s0 <= q0 + 15) {   // wave-uniform: EXEC all-ones inside (WMMA-safe)
      // ---- S = Q K^T (2 key sub-blocks of 16) ----
      v8f sacc[2];
#pragma unroll
      for (int s = 0; s < 2; s++)
#pragma unroll
        for (int e = 0; e < 8; e++) sacc[s][e] = 0.0f;
#pragma unroll
      for (int sub = 0; sub < 2; sub++) {
#pragma unroll
        for (int ks = 0; ks < 2; ks++) {
          Frag bk;
          size_t row = (size_t)(b * TT + s0 + sub * 16 + ll);
          const __bf16* p = Km + row * CC + h * KDIM + ks * 32 + lg * 16;
          bk.q[0] = *(const uint4*)p;
          bk.q[1] = *(const uint4*)(p + 8);
          sacc[sub] = __builtin_amdgcn_wmma_f32_16x16x32_bf16(
              false, aq[ks].v, false, bk.v, (short)0, sacc[sub], false, false);
        }
      }
      // ---- scale + causal mask ----
      float sv[2][8];
#pragma unroll
      for (int sub = 0; sub < 2; sub++)
#pragma unroll
        for (int r = 0; r < 8; r++) {
          int rg = q0 + r + 8 * lg;
          int kg = s0 + sub * 16 + ll;
          float v = sacc[sub][r] * 0.125f;        // 1/sqrt(64)
          sv[sub][r] = (kg > rg) ? -1e30f : v;
        }
      // ---- online softmax (row stats replicated across 16-lane halves) ----
#pragma unroll
      for (int r = 0; r < 8; r++) {
        float mn = fmaxf(sv[0][r], sv[1][r]);
#pragma unroll
        for (int off = 8; off >= 1; off >>= 1)
          mn = fmaxf(mn, __shfl_xor(mn, off, 32));
        float mt    = fmaxf(mrow[r], mn);
        float alpha = __expf(mrow[r] - mt);
        float p0 = __expf(sv[0][r] - mt);
        float p1 = __expf(sv[1][r] - mt);
        sv[0][r] = p0; sv[1][r] = p1;
        float s = p0 + p1;
#pragma unroll
        for (int off = 8; off >= 1; off >>= 1)
          s += __shfl_xor(s, off, 32);
        lrow[r] = lrow[r] * alpha + s;
        mrow[r] = mt;
#pragma unroll
        for (int ni = 0; ni < 4; ni++) o[ni][r] *= alpha;
      }
      // ---- P -> per-wave LDS scratch -> A fragment ----
      unsigned short* pp = &shP[w * 512];
#pragma unroll
      for (int sub = 0; sub < 2; sub++)
#pragma unroll
        for (int r = 0; r < 8; r++) {
          int m   = r + 8 * lg;
          int off = (m + 16 * sub) * 16 + ll;
          pp[off] = f2bfu(sv[sub][r]);
        }
      Frag pf;
      pf.q[0] = *(const uint4*)&pp[lane * 16];
      pf.q[1] = *(const uint4*)&pp[lane * 16 + 8];
      // ---- O += P * V ----
#pragma unroll
      for (int ni = 0; ni < 4; ni++) {
        Frag bv;
        bv.q[0] = *(const uint4*)&shV[ni * 512 + lane * 16];
        bv.q[1] = *(const uint4*)&shV[ni * 512 + lane * 16 + 8];
        o[ni] = __builtin_amdgcn_wmma_f32_16x16x32_bf16(
            false, pf.v, false, bv.v, (short)0, o[ni], false, false);
      }
    }
    __syncthreads();
  }
  // ---- finalize: O /= l, head-concat layout [token, h*64+d] ----
#pragma unroll
  for (int ni = 0; ni < 4; ni++)
#pragma unroll
    for (int r = 0; r < 8; r++) {
      int row = b * TT + q0 + r + 8 * lg;
      int col = h * KDIM + ni * 16 + ll;
      Om[(size_t)row * CC + col] = f2bf(o[ni][r] / lrow[r]);
    }
}

// ---------------------------------------------------------------------------
// LayerNorm(x + y) * g + be  -> f32 + bf16.  One wave per row, shfl-only.
// ---------------------------------------------------------------------------
__global__ __launch_bounds__(256) void k_ln(
    const float* __restrict__ x, const float* __restrict__ y,
    const float* __restrict__ g, const float* __restrict__ be,
    float* __restrict__ xo, __bf16* __restrict__ xbo)
{
  int w    = threadIdx.x >> 5;
  int lane = threadIdx.x & 31;
  int row  = blockIdx.x * 8 + w;
  const float* px = x + (size_t)row * CC;
  const float* py = y ? y + (size_t)row * CC : nullptr;
  int i0 = lane * 16;
  float r[16];
  float s = 0.0f;
#pragma unroll
  for (int j = 0; j < 16; j++) {
    float v = px[i0 + j] + (py ? py[i0 + j] : 0.0f);
    r[j] = v; s += v;
  }
#pragma unroll
  for (int off = 16; off >= 1; off >>= 1) s += __shfl_xor(s, off, 32);
  float mean = s * (1.0f / CC);
  float s2 = 0.0f;
#pragma unroll
  for (int j = 0; j < 16; j++) { float d = r[j] - mean; s2 += d * d; }
#pragma unroll
  for (int off = 16; off >= 1; off >>= 1) s2 += __shfl_xor(s2, off, 32);
  float rstd = rsqrtf(s2 * (1.0f / CC) + 1e-5f);
#pragma unroll
  for (int j = 0; j < 16; j++) {
    int c = i0 + j;
    float v = (r[j] - mean) * rstd * g[c] + be[c];
    xo[(size_t)row * CC + c]  = v;
    xbo[(size_t)row * CC + c] = f2bf(v);
  }
}

// --------------------------- small helper kernels ---------------------------
__global__ void k_cvt(const float* __restrict__ s, __bf16* __restrict__ d, int n) {
  int i = blockIdx.x * blockDim.x + threadIdx.x;
  if (i < n) d[i] = f2bf(s[i]);
}
// Wq/Wk/Wv: [L,H,C,K] -> bf16 [L,C,H*K]
__global__ void k_cvt_qkvT(const float* __restrict__ s, __bf16* __restrict__ d) {
  int idx = blockIdx.x * blockDim.x + threadIdx.x;
  if (idx >= LL * CC * HH * KDIM) return;
  int l   = idx / (CC * HH * KDIM);
  int rem = idx % (CC * HH * KDIM);
  int c   = rem / (HH * KDIM);
  int j   = rem % (HH * KDIM);
  int h   = j / KDIM, k = j % KDIM;
  d[idx] = f2bf(s[(size_t)l * HH * CC * KDIM + (size_t)h * CC * KDIM + c * KDIM + k]);
}
__global__ void k_gather_act(const int* __restrict__ action,
                             const float* __restrict__ Eact,
                             __bf16* __restrict__ cat) {
  int idx = blockIdx.x * blockDim.x + threadIdx.x;
  if (idx >= NTOK * CC) return;
  int row = idx >> 9, c = idx & 511;
  cat[(size_t)row * 1024 + 512 + c] = f2bf(Eact[(size_t)action[row] * CC + c]);
}
__global__ void k_add_time(const float* __restrict__ tok, const int* __restrict__ step,
                           const float* __restrict__ Etime,
                           float* __restrict__ x, __bf16* __restrict__ xb) {
  int idx = blockIdx.x * blockDim.x + threadIdx.x;
  if (idx >= NTOK * CC) return;
  int row = idx >> 9, c = idx & 511;
  float v = tok[idx] + Etime[(size_t)step[row] * CC + c];
  x[idx] = v; xb[idx] = f2bf(v);
}
// Heads: tiny N (128+41+1) -> plain VALU dot kernel, one block per token.
__global__ __launch_bounds__(256) void k_heads(
    const float* __restrict__ x,
    const float* __restrict__ Woh, const float* __restrict__ boh,
    const float* __restrict__ Wrh, const float* __restrict__ brh,
    const float* __restrict__ Wdh, const float* __restrict__ bdh,
    float* __restrict__ out)
{
  __shared__ float sx[CC];
  int row = blockIdx.x;
  for (int i = threadIdx.x; i < CC; i += 256) sx[i] = x[(size_t)row * CC + i];
  __syncthreads();
  int col = threadIdx.x;
  if (col >= 170) return;
  const float* wp; int ld; float bv; float* op;
  if (col < 128)      { wp = Woh + col;          ld = 128; bv = boh[col];       op = out + (size_t)row * 128 + col; }
  else if (col < 169) { int c = col - 128; wp = Wrh + c; ld = 41; bv = brh[c];  op = out + (size_t)NTOK * 128 + (size_t)row * 41 + c; }
  else                { wp = Wdh;                ld = 1;   bv = bdh[0];         op = out + (size_t)NTOK * 128 + (size_t)NTOK * 41 + row; }
  float s = bv;
  for (int c = 0; c < CC; c++) s += sx[c] * wp[(size_t)c * ld];
  *op = s;
}

// ---------------------------------------------------------------------------
extern "C" void kernel_launch(void* const* d_in, const int* in_sizes, int n_in,
                              void* d_out, int out_size, void* d_ws, size_t ws_size,
                              hipStream_t stream) {
  const float* obs    = (const float*)d_in[0];
  const int*   action = (const int*)  d_in[1];
  const int*   step   = (const int*)  d_in[2];
  const float* W_obs  = (const float*)d_in[3];
  const float* b_obs  = (const float*)d_in[4];
  const float* E_act  = (const float*)d_in[5];
  const float* W_traj = (const float*)d_in[6];
  const float* b_traj = (const float*)d_in[7];
  const float* E_time = (const float*)d_in[8];
  const float* Wq     = (const float*)d_in[9];
  const float* Wk     = (const float*)d_in[10];
  const float* Wv     = (const float*)d_in[11];
  const float* Wp     = (const float*)d_in[12];
  const float* bp     = (const float*)d_in[13];
  const float* W1     = (const float*)d_in[14];
  const float* b1     = (const float*)d_in[15];
  const float* W2     = (const float*)d_in[16];
  const float* b2     = (const float*)d_in[17];
  const float* g1v    = (const float*)d_in[18];
  const float* be1    = (const float*)d_in[19];
  const float* g2v    = (const float*)d_in[20];
  const float* be2    = (const float*)d_in[21];
  const float* gfv    = (const float*)d_in[22];
  const float* bfv    = (const float*)d_in[23];
  const float* W_oh   = (const float*)d_in[24];
  const float* b_oh   = (const float*)d_in[25];
  const float* W_rh   = (const float*)d_in[26];
  const float* b_rh   = (const float*)d_in[27];
  const float* W_dh   = (const float*)d_in[28];
  const float* b_dh   = (const float*)d_in[29];

  // ---- carve workspace ----
  char* p = (char*)d_ws;
  auto carve = [&](size_t bytes) -> char* {
    char* r = p; p += (bytes + 255) & ~(size_t)255; return r;
  };
  __bf16* wObsB  = (__bf16*)carve((size_t)128 * 512 * 2);
  __bf16* wTrajB = (__bf16*)carve((size_t)1024 * 512 * 2);
  __bf16* wQB    = (__bf16*)carve((size_t)LL * 512 * 512 * 2);
  __bf16* wKB    = (__bf16*)carve((size_t)LL * 512 * 512 * 2);
  __bf16* wVB    = (__bf16*)carve((size_t)LL * 512 * 512 * 2);
  __bf16* wPB    = (__bf16*)carve((size_t)LL * 512 * 512 * 2);
  __bf16* w1B    = (__bf16*)carve((size_t)LL * 512 * 2048 * 2);
  __bf16* w2B    = (__bf16*)carve((size_t)LL * 2048 * 512 * 2);
  float*  xf     = (float*) carve((size_t)NTOK * CC * 4);
  __bf16* xb     = (__bf16*)carve((size_t)NTOK * CC * 2);
  __bf16* qB     = (__bf16*)carve((size_t)NTOK * CC * 2);
  __bf16* kB     = (__bf16*)carve((size_t)NTOK * CC * 2);
  __bf16* vB     = (__bf16*)carve((size_t)NTOK * CC * 2);
  __bf16* oB     = (__bf16*)carve((size_t)NTOK * CC * 2);   // also obs_bf16 in embed
  float*  ybuf   = (float*) carve((size_t)NTOK * CC * 4);
  __bf16* big    = (__bf16*)carve((size_t)NTOK * FFD * 2);  // h1 / concat buffer

  auto g1d = [](int n) { return dim3((unsigned)((n + 255) / 256)); };
  const dim3 blk(256);

  // ---- weight conversion ----
  k_cvt<<<g1d(128 * 512), blk, 0, stream>>>(W_obs, wObsB, 128 * 512);
  k_cvt<<<g1d(1024 * 512), blk, 0, stream>>>(W_traj, wTrajB, 1024 * 512);
  k_cvt_qkvT<<<g1d(LL * 512 * 512), blk, 0, stream>>>(Wq, wQB);
  k_cvt_qkvT<<<g1d(LL * 512 * 512), blk, 0, stream>>>(Wk, wKB);
  k_cvt_qkvT<<<g1d(LL * 512 * 512), blk, 0, stream>>>(Wv, wVB);
  k_cvt<<<g1d(LL * 512 * 512), blk, 0, stream>>>(Wp, wPB, LL * 512 * 512);
  k_cvt<<<g1d(LL * 512 * 2048), blk, 0, stream>>>(W1, w1B, LL * 512 * 2048);
  k_cvt<<<g1d(LL * 2048 * 512), blk, 0, stream>>>(W2, w2B, LL * 2048 * 512);

  // ---- embeddings ----
  k_cvt<<<g1d(NTOK * 128), blk, 0, stream>>>(obs, oB, NTOK * 128);  // obs -> bf16
  const dim3 gC(512 / 128, NTOK / 128);       // (4, 64)
  const dim3 gF(2048 / 128, NTOK / 128);      // (16, 64)
  // obs_emb (bf16) into concat cols [0,512)
  k_gemm<<<gC, blk, 0, stream>>>(oB, 128, wObsB, 512, nullptr, 0, big, 1024, b_obs, 0, 128);
  // act_emb gather into concat cols [512,1024)
  k_gather_act<<<g1d(NTOK * CC), blk, 0, stream>>>(action, E_act, big);
  // token = concat @ W_traj + b_traj
  k_gemm<<<gC, blk, 0, stream>>>(big, 1024, wTrajB, 512, ybuf, 512, nullptr, 0, b_traj, 0, 1024);
  // x = token + E_time[step]
  k_add_time<<<g1d(NTOK * CC), blk, 0, stream>>>(ybuf, step, E_time, xf, xb);

  // ---- transformer layers ----
  for (int l = 0; l < LL; l++) {
    const size_t wOff = (size_t)l * 512 * 512;
    k_gemm<<<gC, blk, 0, stream>>>(xb, 512, wQB + wOff, 512, nullptr, 0, qB, 512, nullptr, 0, 512);
    k_gemm<<<gC, blk, 0, stream>>>(xb, 512, wKB + wOff, 512, nullptr, 0, kB, 512, nullptr, 0, 512);
    k_gemm<<<gC, blk, 0, stream>>>(xb, 512, wVB + wOff, 512, nullptr, 0, vB, 512, nullptr, 0, 512);
    k_attn<<<dim3(TT / 64, HH, BB), dim3(128), 0, stream>>>(qB, kB, vB, oB);
    k_gemm<<<gC, blk, 0, stream>>>(oB, 512, wPB + wOff, 512, ybuf, 512, nullptr, 0, bp + l * 512, 0, 512);
    k_ln<<<dim3(NTOK / 8), blk, 0, stream>>>(xf, ybuf, g1v + l * 512, be1 + l * 512, xf, xb);
    k_gemm<<<gF, blk, 0, stream>>>(xb, 512, w1B + (size_t)l * 512 * 2048, 2048,
                                   nullptr, 0, big, 2048, b1 + l * 2048, 1, 512);
    k_gemm<<<gC, blk, 0, stream>>>(big, 2048, w2B + (size_t)l * 2048 * 512, 512,
                                   ybuf, 512, nullptr, 0, b2 + l * 512, 0, 2048);
    k_ln<<<dim3(NTOK / 8), blk, 0, stream>>>(xf, ybuf, g2v + l * 512, be2 + l * 512, xf, xb);
  }

  // ---- final norm + heads ----
  k_ln<<<dim3(NTOK / 8), blk, 0, stream>>>(xf, nullptr, gfv, bfv, xf, xb);
  k_heads<<<dim3(NTOK), blk, 0, stream>>>(xf, W_oh, b_oh, W_rh, b_rh, W_dh, b_dh,
                                          (float*)d_out);
}
